// Gen2AssocModel_27917287424322
// MI455X (gfx1250) — matrix operand order repaired and checked by reading, not compiled, not run on hardware
//
#include <hip/hip_runtime.h>
#include <hip/hip_bf16.h>

// ---------------------------------------------------------------------------
// CDNA5 (gfx1250) leaky associative spiking model.
//   - All GEMM operands pre-converted once to bf16 (weights transposed+padded):
//     GEMM hot loop is pure async global->LDS b128 copies + WMMA.
//   - Double-buffered GLOBAL_LOAD_ASYNC_TO_LDS_B128 (ASYNCcnt) pipelining:
//     slab k+1 copies overlap the 8x v_wmma_f32_16x16x32_bf16 of slab k.
//   - k/v/q fused into one N=96 GEMM; recurrent scan parallel over B*N*N.
// ---------------------------------------------------------------------------

typedef __attribute__((ext_vector_type(16))) __bf16 v16bf;
typedef __attribute__((ext_vector_type(8)))  float  v8f;

__device__ __forceinline__ unsigned short bf16u(float f) {
    __bf16 h = (__bf16)f;
    unsigned short u; __builtin_memcpy(&u, &h, 2);
    return u;
}

// 8 contiguous LDS dwords -> one 16-element bf16 fragment (ds_load_b128 x2)
__device__ __forceinline__ v16bf frag_from_lds(const unsigned* p) {
    union { v16bf v; unsigned u[8]; } t;
    #pragma unroll
    for (int i = 0; i < 8; ++i) t.u[i] = p[i];
    return t.v;
}

// LDS geometry: one shared block at LDS offset 0, two slab buffers.
// Each buffer: A rows 128 x 20 dwords, then B rows 128 x 20 dwords.
#define LDS_ROW_DW   20
#define LDS_B_BASEDW (128 * LDS_ROW_DW)          // 2560
#define LDS_BUF_DW   (2 * 128 * LDS_ROW_DW)      // 5120 dwords = 20 KB

// ---------------------------------------------------------------------------
// Tiled WMMA GEMM:  C[M,Nn] = act( A[M,Kpad]_bf16 * Bt[Npad,Kpad]_bf16^T + bias )
// 256 threads = 8 waves (4M x 2N), block tile 128x128, K slab 32.
// A rows in LDS carry the K 8-group permutation {0,2,1,3} so lane-half h reads
// dwords [h*8, h*8+8); Bt rows carry the same permutation (K-major per output column).
// Per wave each slab issues exactly 4 async instructions, so after issuing the
// next slab, s_wait_asynccnt 4 == "previous slab complete, next in flight".
// No bounds checks: M % 128 == 0, Kpad % 32 == 0, Npad >= gridDim.x*128.
// ACT: 0 none | 1 relu | 2 sigmoid | 3 sigmoid for cols >= sigFrom
// OUTBF: write bf16 (feeds another GEMM) vs f32 (feeds scan / final output)
// ---------------------------------------------------------------------------
template <int ACT, bool OUTBF>
__global__ __launch_bounds__(256)
void gemm_bf16_wmma(const unsigned short* __restrict__ A,
                    const unsigned short* __restrict__ Bt,
                    const float* __restrict__ bias, void* __restrict__ Cout,
                    int M, int Nn, int Kpad, int sigFrom) {
    __shared__ unsigned lds[2 * LDS_BUF_DW];     // 40 KB (double buffer)

    const int blockM = blockIdx.y * 128;
    const int blockN = blockIdx.x * 128;

    const int lane  = threadIdx.x & 31;
    const int wave  = threadIdx.x >> 5;
    const int waveM = wave >> 1;             // 0..3 -> 32-row slab
    const int waveN = wave & 1;              // 0..1 -> 64-col slab
    const int half  = lane >> 4;
    const int r16   = lane & 15;

    // issue one slab's async copies (4 async instructions per thread/wave)
    auto issue_slab = [&](int slab, int buf) {
        const int kb = slab * 32;
        const unsigned bufB = (unsigned)(buf * LDS_BUF_DW);
        #pragma unroll
        for (int it = 0; it < 2; ++it) {
            const int chunk = threadIdx.x + it * 256;   // 0..511
            const int rr = chunk >> 2;                  // tile row / tile col
            const int g  = chunk & 3;                   // 8-element K group
            const int pg = ((g & 1) << 1) | (g >> 1);   // perm {0,2,1,3}
            {
                const unsigned long long src = (unsigned long long)
                    (A + (size_t)(blockM + rr) * Kpad + kb + g * 8);
                const unsigned dst = (bufB + rr * LDS_ROW_DW + pg * 4) * 4;
                asm volatile("global_load_async_to_lds_b128 %0, %1, off"
                             :: "v"(dst), "v"(src) : "memory");
            }
            {
                const unsigned long long src = (unsigned long long)
                    (Bt + (size_t)(blockN + rr) * Kpad + kb + g * 8);
                const unsigned dst =
                    (bufB + LDS_B_BASEDW + rr * LDS_ROW_DW + pg * 4) * 4;
                asm volatile("global_load_async_to_lds_b128 %0, %1, off"
                             :: "v"(dst), "v"(src) : "memory");
            }
        }
    };

    v8f acc[2][4];
    #pragma unroll
    for (int r = 0; r < 2; ++r)
        #pragma unroll
        for (int c = 0; c < 4; ++c)
            acc[r][c] = (v8f){0.f,0.f,0.f,0.f,0.f,0.f,0.f,0.f};

    const int nslab = Kpad >> 5;
    issue_slab(0, 0);

    for (int s = 0; s < nslab; ++s) {
        const int cur = s & 1;
        if (s + 1 < nslab) {
            issue_slab(s + 1, cur ^ 1);                       // prefetch overlap
            asm volatile("s_wait_asynccnt 4" ::: "memory");   // slab s done
        } else {
            asm volatile("s_wait_asynccnt 0" ::: "memory");
        }
        __syncthreads();

        // ---- fragments (ds_load_b128) + 2x4 WMMAs on buffer `cur` ----
        const unsigned* buf = &lds[cur * LDS_BUF_DW];
        v16bf afrag[2], bfrag[4];
        #pragma unroll
        for (int r = 0; r < 2; ++r)
            afrag[r] = frag_from_lds(
                &buf[(waveM * 32 + r * 16 + r16) * LDS_ROW_DW + half * 8]);
        #pragma unroll
        for (int c = 0; c < 4; ++c)
            bfrag[c] = frag_from_lds(
                &buf[LDS_B_BASEDW + (waveN * 64 + c * 16 + r16) * LDS_ROW_DW + half * 8]);
        #pragma unroll
        for (int r = 0; r < 2; ++r)
            #pragma unroll
            for (int c = 0; c < 4; ++c)
                acc[r][c] = __builtin_amdgcn_wmma_f32_16x16x32_bf16(
                    false, afrag[r], false, bfrag[c],
                    (short)0, acc[r][c], false, false);
        __syncthreads();   // protects buffer overwritten by next issue
    }

    // ---- epilogue: bias + activation + store ----
    #pragma unroll
    for (int r = 0; r < 2; ++r) {
        #pragma unroll
        for (int c = 0; c < 4; ++c) {
            const int gn = blockN + waveN * 64 + c * 16 + r16;
            if (gn >= Nn) continue;                       // padded columns
            const int gmBase = blockM + waveM * 32 + r * 16 + half * 8;
            const float bval = bias[gn];
            #pragma unroll
            for (int vr = 0; vr < 8; ++vr) {
                const int gm = gmBase + vr;
                float v = acc[r][c][vr] + bval;
                if (ACT == 1) v = v > 0.0f ? v : 0.0f;
                else if (ACT == 2) v = 1.0f / (1.0f + __expf(-v));
                else if (ACT == 3) { if (gn >= sigFrom) v = 1.0f / (1.0f + __expf(-v)); }
                if (OUTBF)
                    ((unsigned short*)Cout)[(size_t)gm * Nn + gn] = bf16u(v);
                else
                    ((float*)Cout)[(size_t)gm * Nn + gn] = v;
            }
        }
    }
}

// ---------------------------------------------------------------------------
// Prep kernels (run once per launch; tiny vs the GEMMs)
// ---------------------------------------------------------------------------

// x (M x 65 f32) -> xb (M x 96 bf16, zero-padded K tail)
__global__ void conv_x_bf16(const float* __restrict__ x,
                            unsigned short* __restrict__ xb, int M) {
    const int idx = blockIdx.x * blockDim.x + threadIdx.x;
    if (idx >= M * 96) return;
    const int m = idx / 96, k = idx % 96;
    xb[idx] = bf16u(k < 65 ? x[(size_t)m * 65 + k] : 0.0f);
}

// W (K x N f32) -> Wt (Npad x Kpad bf16), transposed + zero-padded
__global__ void transpose_w_bf16(const float* __restrict__ W,
                                 unsigned short* __restrict__ Wt,
                                 int K, int N, int Kpad, int Npad) {
    const int idx = blockIdx.x * blockDim.x + threadIdx.x;
    if (idx >= Npad * Kpad) return;
    const int n = idx / Kpad, k = idx % Kpad;
    const float v = (n < N && k < K) ? W[(size_t)k * N + n] : 0.0f;
    Wt[idx] = bf16u(v);
}

// [Wk|Wv|Wq] (each K x 32 f32) -> Wt (128 x K bf16, rows 96..127 zero) + bias96
__global__ void pack_kvq_T(const float* __restrict__ Wk, const float* __restrict__ Wv,
                           const float* __restrict__ Wq,
                           const float* __restrict__ bk, const float* __restrict__ bv,
                           const float* __restrict__ bq,
                           unsigned short* __restrict__ Wt, float* __restrict__ bp,
                           int K) {
    const int idx = blockIdx.x * blockDim.x + threadIdx.x;
    if (idx < 128 * K) {
        const int n = idx / K, k = idx % K;
        float v = 0.0f;
        if (n < 96) {
            const int j = n >> 5, c = n & 31;
            const float* W = (j == 0) ? Wk : (j == 1) ? Wv : Wq;
            v = W[(size_t)k * 32 + c];
        }
        Wt[idx] = bf16u(v);
    }
    if (idx < 32) { bp[idx] = bk[idx]; bp[32 + idx] = bv[idx]; bp[64 + idx] = bq[idx]; }
}

// ---------------------------------------------------------------------------
// Recurrent associative scan over packed kvq ((T*B) x 96 f32, q pre-sigmoided).
// One block per batch element; thread (i,j) holds mem[i][j] for all T steps.
// Output written as bf16 (only ever consumed by GEMMs).
// ---------------------------------------------------------------------------
__global__ __launch_bounds__(1024)
void assoc_scan(const float* __restrict__ kvq, unsigned short* __restrict__ Out,
                int T, int Bt) {
    __shared__ float s[96];                  // [k | v | q]
    const int b   = blockIdx.x;
    const int tid = threadIdx.x;
    const int i   = tid >> 5;
    const int j   = tid & 31;
    float mem = 0.0f;

    for (int t = 0; t < T; ++t) {
        if (tid < 96) s[tid] = kvq[((size_t)t * Bt + b) * 96 + tid];
        __syncthreads();

        mem = 0.9f * mem + s[i] * s[32 + j];
        const float sp = (mem - 1.0f) > 0.0f ? 1.0f : 0.0f;   // THRESH = 1
        mem -= sp;                                            // soft reset
        Out[((size_t)t * Bt + b) * 1024 + tid] = bf16u(sp * s[64 + i]);
        __syncthreads();
    }
}

// ---------------------------------------------------------------------------
// Orchestration
// ---------------------------------------------------------------------------
extern "C" void kernel_launch(void* const* d_in, const int* in_sizes, int n_in,
                              void* d_out, int out_size, void* d_ws, size_t ws_size,
                              hipStream_t stream) {
    (void)in_sizes; (void)n_in; (void)out_size; (void)ws_size;

    const int T = 512, B = 64, HID = 1024, SW = 64;
    const int M = T * B;                       // 32768

    const float* x     = (const float*)d_in[0];
    const float* W_in  = (const float*)d_in[1];
    const float* b_in  = (const float*)d_in[2];
    const float* Wk1   = (const float*)d_in[3];
    const float* bk1   = (const float*)d_in[4];
    const float* Wv1   = (const float*)d_in[5];
    const float* bv1   = (const float*)d_in[6];
    const float* Wq1   = (const float*)d_in[7];
    const float* bq1   = (const float*)d_in[8];
    const float* W_mid = (const float*)d_in[9];
    const float* b_mid = (const float*)d_in[10];
    const float* Wk2   = (const float*)d_in[11];
    const float* bk2   = (const float*)d_in[12];
    const float* Wv2   = (const float*)d_in[13];
    const float* bv2   = (const float*)d_in[14];
    const float* Wq2   = (const float*)d_in[15];
    const float* bq2   = (const float*)d_in[16];
    const float* W_out = (const float*)d_in[17];
    const float* b_out = (const float*)d_in[18];

    // ---- workspace carve (all 16B-aligned offsets) ----
    unsigned short* hAb  = (unsigned short*)d_ws;          // M x 1024 bf16
    unsigned short* hBb  = hAb + (size_t)M * 1024;         // M x 1024 bf16
    unsigned short* xb   = hBb + (size_t)M * 1024;         // M x 96   bf16
    unsigned short* WinT = xb  + (size_t)M * 96;           // 1024 x 96
    unsigned short* WmidT= WinT + (size_t)1024 * 96;       // 1024 x 1024
    unsigned short* Wp1T = WmidT + (size_t)1024 * 1024;    // 128 x 1024
    unsigned short* Wp2T = Wp1T + (size_t)128 * 1024;      // 128 x 1024
    unsigned short* WoutT= Wp2T + (size_t)128 * 1024;      // 128 x 1024
    float* kvqo = (float*)(WoutT + (size_t)128 * 1024);    // M x 96 f32
    float* bp1  = kvqo + (size_t)M * 96;                   // 96 (+pad)
    float* bp2  = bp1 + 128;

    const dim3 blk(256);

    // ---- one-time prep: bf16 conversion / transposition / packing ----
    conv_x_bf16<<<dim3((M * 96 + 255) / 256), blk, 0, stream>>>(x, xb, M);
    transpose_w_bf16<<<dim3((1024 * 96 + 255) / 256), blk, 0, stream>>>(
        W_in, WinT, 65, HID, 96, 1024);
    transpose_w_bf16<<<dim3((1024 * 1024 + 255) / 256), blk, 0, stream>>>(
        W_mid, WmidT, HID, HID, 1024, 1024);
    transpose_w_bf16<<<dim3((128 * 1024 + 255) / 256), blk, 0, stream>>>(
        W_out, WoutT, HID, SW, 1024, 128);
    pack_kvq_T<<<dim3((128 * 1024 + 255) / 256), blk, 0, stream>>>(
        Wk1, Wv1, Wq1, bk1, bv1, bq1, Wp1T, bp1, HID);
    pack_kvq_T<<<dim3((128 * 1024 + 255) / 256), blk, 0, stream>>>(
        Wk2, Wv2, Wq2, bk2, bv2, bq2, Wp2T, bp2, HID);

    const dim3 gridHID(HID / 128, M / 128);
    const dim3 grid1(1, M / 128);

    // h0 = x @ W_in + b_in  (bf16 out)
    gemm_bf16_wmma<0, true><<<gridHID, blk, 0, stream>>>(
        xb, WinT, b_in, hAb, M, HID, 96, 0);

    // layer 1: fused kvq (q sigmoided, cols >= 64), then scan
    gemm_bf16_wmma<3, false><<<grid1, blk, 0, stream>>>(
        hAb, Wp1T, bp1, kvqo, M, 96, HID, 64);
    assoc_scan<<<dim3(B), dim3(1024), 0, stream>>>(kvqo, hBb, T, B);

    // h_mid = relu(hB @ W_mid + b_mid)  (bf16 out)
    gemm_bf16_wmma<1, true><<<gridHID, blk, 0, stream>>>(
        hBb, WmidT, b_mid, hAb, M, HID, HID, 0);

    // layer 2: fused kvq, then scan
    gemm_bf16_wmma<3, false><<<grid1, blk, 0, stream>>>(
        hAb, Wp2T, bp2, kvqo, M, 96, HID, 64);
    assoc_scan<<<dim3(B), dim3(1024), 0, stream>>>(kvqo, hBb, T, B);

    // y = sigmoid(hB @ W_out + b_out) -> d_out (f32)
    gemm_bf16_wmma<2, false><<<grid1, blk, 0, stream>>>(
        hBb, WoutT, b_out, (float*)d_out, M, SW, HID, 0);
}
